// TransformerBlock_39926015984079
// MI455X (gfx1250) — matrix operand order repaired and checked by reading, not compiled, not run on hardware
//
#include <hip/hip_runtime.h>
#include <math.h>
#include <stdint.h>

// ---------------------------------------------------------------------------
// CDNA5 (gfx1250, wave32) transformer block forward.
// GEMMs: v_wmma_f32_16x16x32_bf16 (fp32 accumulate), weights staged in LDS
// via global_load_async_to_lds_b128 (ASYNCcnt) with bank-conflict-free pads.
// ---------------------------------------------------------------------------

typedef __attribute__((ext_vector_type(16))) __bf16 v16bf;
typedef __attribute__((ext_vector_type(8)))  __bf16 v8bf;
typedef __attribute__((ext_vector_type(8)))  float  v8f;

union Frag16 { v16bf v; v8bf h[2]; };

enum { EPI_BF16 = 0, EPI_GELU = 1, EPI_RESF32 = 2 };

__device__ __forceinline__ float gelu_f(float x) {
  const float c = 0.7978845608028654f;  // sqrt(2/pi)
  float u = c * (x + 0.044715f * x * x * x);
  return 0.5f * x * (1.0f + tanhf(u));
}

// One per-lane async global->LDS 16-byte transfer (gfx1250, ASYNCcnt).
// lds_off: LDS byte offset (wave-relative), g: per-lane global address.
__device__ __forceinline__ void async_copy_b128(uint32_t lds_off, const __bf16* g) {
  asm volatile("global_load_async_to_lds_b128 %0, %1, off"
               :: "v"(lds_off), "v"(g) : "memory");
}
__device__ __forceinline__ void wait_async0() {
  asm volatile("s_wait_asynccnt 0x0" ::: "memory");
}

// ---------------------------------------------------------------------------
// fp32 [R][C] -> bf16 [C][R]  (weights pre-transposed so WMMA B-fragments are
// contiguous: lane = output column N, K = 8h..8h+7 and 16+8h..16+8h+7 per lane).
// ---------------------------------------------------------------------------
__global__ __launch_bounds__(256) void transpose_to_bf16(
    const float* __restrict__ src, __bf16* __restrict__ dst, int R, int C) {
  __shared__ float tile[32][33];
  const int c0 = blockIdx.x * 32;
  const int r0 = blockIdx.y * 32;
  const int tx = threadIdx.x;  // 0..31
  const int ty = threadIdx.y;  // 0..7
#pragma unroll
  for (int j = 0; j < 32; j += 8)
    tile[ty + j][tx] = src[(size_t)(r0 + ty + j) * C + (c0 + tx)];
  __syncthreads();
#pragma unroll
  for (int j = 0; j < 32; j += 8)
    dst[(size_t)(c0 + ty + j) * R + (r0 + tx)] = (__bf16)tile[tx][ty + j];
}

// ---------------------------------------------------------------------------
// Row LayerNorm (E = 1024), fp32 stats, bf16 output (GEMM A operand).
// ---------------------------------------------------------------------------
__global__ __launch_bounds__(256) void layernorm_to_bf16(
    const float* __restrict__ x, const float* __restrict__ g,
    const float* __restrict__ bb, __bf16* __restrict__ y, int E) {
  __shared__ float red[256];
  const int row = blockIdx.x;
  const int tid = threadIdx.x;
  const float* xr = x + (size_t)row * E;

  float v4[4];
  float s = 0.f;
#pragma unroll
  for (int i = 0; i < 4; ++i) { v4[i] = xr[tid + i * 256]; s += v4[i]; }
  red[tid] = s;
  __syncthreads();
  for (int k = 128; k > 0; k >>= 1) {
    if (tid < k) red[tid] += red[tid + k];
    __syncthreads();
  }
  const float mu = red[0] * (1.0f / 1024.0f);
  __syncthreads();

  s = 0.f;
#pragma unroll
  for (int i = 0; i < 4; ++i) { float d = v4[i] - mu; s += d * d; }
  red[tid] = s;
  __syncthreads();
  for (int k = 128; k > 0; k >>= 1) {
    if (tid < k) red[tid] += red[tid + k];
    __syncthreads();
  }
  const float rstd = rsqrtf(red[0] * (1.0f / 1024.0f) + 1e-5f);

#pragma unroll
  for (int i = 0; i < 4; ++i) {
    const int c = tid + i * 256;
    y[(size_t)row * E + c] = (__bf16)((v4[i] - mu) * rstd * g[c] + bb[c]);
  }
}

// ---------------------------------------------------------------------------
// WMMA GEMM: Y[M][N] = A[M][K] * Bt[N][K]^T + bias, fused epilogue.
// Block = 256 threads = 8 waves (4 M x 2 N); wave tile 32x32; block 128x64.
// B sub-tile (64 x KCHUNK) staged in LDS via async-to-LDS; padded rows so
// 16-lane ds_load_b128 fragment reads hit all 64 banks.
// ---------------------------------------------------------------------------
#define KCHUNK 256
#define BROW   (KCHUNK + 8)   // halves; 132 DWORDs/row => 4-bank row skew

template <int EPI>
__global__ __launch_bounds__(256) void gemm_bf16_wmma(
    const __bf16* __restrict__ A,     // [M][K] bf16 row-major
    const __bf16* __restrict__ Bt,    // [N][K] bf16 (transposed weights)
    const float*  __restrict__ bias,  // [N]
    const float*  __restrict__ res,   // [M][N] residual (EPI_RESF32)
    __bf16*       __restrict__ outb,  // bf16 output
    float*        __restrict__ outf,  // f32 output
    int M, int N, int K) {
  __shared__ alignas(16) __bf16 Bs[64][BROW];  // ~33.8 KB

  const int tid  = threadIdx.x;
  const int lane = tid & 31;
  const int wave = tid >> 5;
  const int lr = lane & 15;   // row (A) / col (B,C) within 16x16 tile
  const int lh = lane >> 4;   // half selector
  const int wm = wave & 3;
  const int wn = wave >> 2;
  const int m0 = blockIdx.x * 128 + wm * 32;
  const int nblk = blockIdx.y * 64;

  v8f acc[2][2] = {};

  const __bf16* pa0 = A + (size_t)(m0 + lr) * K + lh * 8;
  const __bf16* pa1 = pa0 + (size_t)16 * K;

  const uint32_t bs_base = (uint32_t)(uintptr_t)&Bs[0][0];
  // this thread's 8 async transfers per chunk: transfer idx = tid + i*256,
  // row = idx>>5 (0..63), seg = idx&31 (16B each; 32 segs * 16B = 512B row)
  const __bf16* bsrc0 = Bt + (size_t)(nblk + (tid >> 5) * 8) * K;  // unused helper
  (void)bsrc0;

  for (int kc = 0; kc < K; kc += KCHUNK) {
    // ---- cooperative async copy: Bs[r][0..255] = Bt[nblk+r][kc..kc+255] ----
#pragma unroll
    for (int i = 0; i < 8; ++i) {
      const int idx = tid + i * 256;       // 0..2047
      const int r   = idx >> 5;            // 0..63
      const int seg = idx & 31;            // 0..31 (8 halves each)
      async_copy_b128(bs_base + (uint32_t)(r * (BROW * 2) + seg * 16),
                      Bt + (size_t)(nblk + r) * K + kc + seg * 8);
    }
    wait_async0();
    __syncthreads();

    // ---- 8 WMMA k-steps over the chunk ----
#pragma unroll
    for (int ks = 0; ks < KCHUNK; ks += 32) {
      const int k0 = kc + ks;
      Frag16 a0, a1;
      a0.h[0] = *(const v8bf*)(pa0 + k0);       // K = lh*8 .. +7
      a0.h[1] = *(const v8bf*)(pa0 + k0 + 16);  // K = 16+lh*8 .. +7
      a1.h[0] = *(const v8bf*)(pa1 + k0);
      a1.h[1] = *(const v8bf*)(pa1 + k0 + 16);
      Frag16 fb0, fb1;
      fb0.h[0] = *(const v8bf*)&Bs[wn * 32 + lr][ks + lh * 8];
      fb0.h[1] = *(const v8bf*)&Bs[wn * 32 + lr][ks + 16 + lh * 8];
      fb1.h[0] = *(const v8bf*)&Bs[wn * 32 + 16 + lr][ks + lh * 8];
      fb1.h[1] = *(const v8bf*)&Bs[wn * 32 + 16 + lr][ks + 16 + lh * 8];
      v16bf b0 = fb0.v;
      v16bf b1 = fb1.v;
      __builtin_prefetch(pa0 + k0 + 512, 0, 1);  // global_prefetch_b8
      acc[0][0] = __builtin_amdgcn_wmma_f32_16x16x32_bf16(false, a0.v, false, b0, (short)0, acc[0][0], false, false);
      acc[0][1] = __builtin_amdgcn_wmma_f32_16x16x32_bf16(false, a0.v, false, b1, (short)0, acc[0][1], false, false);
      acc[1][0] = __builtin_amdgcn_wmma_f32_16x16x32_bf16(false, a1.v, false, b0, (short)0, acc[1][0], false, false);
      acc[1][1] = __builtin_amdgcn_wmma_f32_16x16x32_bf16(false, a1.v, false, b1, (short)0, acc[1][1], false, false);
    }
    __syncthreads();  // before next chunk overwrites Bs
  }

  // C layout: lane lr = column, rows (lh*8 + r), r = VGPR index 0..7
#pragma unroll
  for (int mi = 0; mi < 2; ++mi) {
#pragma unroll
    for (int ni = 0; ni < 2; ++ni) {
      const int row = m0 + mi * 16 + lh * 8;
      const int col = nblk + wn * 32 + ni * 16 + lr;
      const float bv = bias[col];
#pragma unroll
      for (int r = 0; r < 8; ++r) {
        float v = acc[mi][ni][r] + bv;
        const size_t idx = (size_t)(row + r) * N + col;
        if constexpr (EPI == EPI_BF16) {
          outb[idx] = (__bf16)v;
        } else if constexpr (EPI == EPI_GELU) {
          outb[idx] = (__bf16)gelu_f(v);
        } else {
          outf[idx] = res[idx] + v;
        }
      }
    }
  }
}

// ---------------------------------------------------------------------------
// Split fused QKV [B,T,3E] (bf16, bias applied) into Qc,Kc [B,H,T,64] and
// Vt [B,H,64,T] so attention WMMA fragments load contiguously.
// ---------------------------------------------------------------------------
__global__ __launch_bounds__(256) void split_qkv(
    const __bf16* __restrict__ qkv, __bf16* __restrict__ Qc,
    __bf16* __restrict__ Kc, __bf16* __restrict__ Vt,
    int B, int T, int E, int H) {
  const size_t i = (size_t)blockIdx.x * 256 + threadIdx.x;
  const size_t total = (size_t)B * T * E;
  if (i >= total) return;
  const int e = (int)(i % E);
  const size_t t = (i / E) % T;
  const size_t b = i / ((size_t)T * E);
  const int h = e >> 6;
  const int d = e & 63;
  const size_t src = (b * T + t) * (size_t)(3 * E) + e;
  const size_t bh = b * H + h;
  const size_t qi = (bh * T + t) * 64 + d;
  Qc[qi] = qkv[src];
  Kc[qi] = qkv[src + E];
  Vt[(bh * 64 + d) * T + t] = qkv[src + 2 * E];
}

// ---------------------------------------------------------------------------
// Flash attention (causal): one wave per 16-row query tile, 32-key blocks.
// S = Q K^T via WMMA; online softmax fp32 with shfl reductions over the
// 16-lane row groups of the C layout; P staged through LDS (C-layout ->
// A-fragment layout); O += P V via WMMA against pre-transposed V.
// ---------------------------------------------------------------------------
__global__ __launch_bounds__(256) void flash_attn(
    const __bf16* __restrict__ Qc, const __bf16* __restrict__ Kc,
    const __bf16* __restrict__ Vt, __bf16* __restrict__ O,
    int T, int H, int E) {
  __shared__ alignas(32) __bf16 pstage[8][16][32];  // per-wave P tile

  const int lane = threadIdx.x & 31;
  const int wave = threadIdx.x >> 5;
  const int lr = lane & 15;
  const int lh = lane >> 4;
  const int h = blockIdx.y;
  const int b = blockIdx.z;
  const int q0 = (blockIdx.x * 8 + wave) * 16;
  const size_t bh = (size_t)b * H + h;

  const __bf16* Qb = Qc + bh * (size_t)T * 64;
  const __bf16* Kb = Kc + bh * (size_t)T * 64;
  const __bf16* Vb = Vt + bh * (size_t)64 * T;

  // Q fragments for D = 0..31 and 32..63 (reused across all key blocks)
  Frag16 aq[2];
  const __bf16* pq = Qb + (size_t)(q0 + lr) * 64 + lh * 16;
  aq[0].h[0] = *(const v8bf*)(pq);
  aq[0].h[1] = *(const v8bf*)(pq + 8);
  aq[1].h[0] = *(const v8bf*)(pq + 32);
  aq[1].h[1] = *(const v8bf*)(pq + 40);

  v8f accO[4] = {};
  float mrow[8], lrow[8];
#pragma unroll
  for (int r = 0; r < 8; ++r) { mrow[r] = -1e30f; lrow[r] = 0.f; }

  const int nkb = (q0 + 16 + 31) >> 5;  // causal: keys <= q0+15
  for (int kb = 0; kb < nkb; ++kb) {
    const int kbase = kb * 32;

    // ---- S = Q K^T (16x32 key block) ----
    v8f s[2] = {};
#pragma unroll
    for (int ni = 0; ni < 2; ++ni) {
      const __bf16* pk = Kb + (size_t)(kbase + ni * 16 + lr) * 64 + lh * 16;
      v16bf bk0 = *(const v16bf*)(pk);       // D = lh*16 .. +15
      v16bf bk1 = *(const v16bf*)(pk + 32);  // D = 32 + lh*16 .. +15
      s[ni] = __builtin_amdgcn_wmma_f32_16x16x32_bf16(false, aq[0].v, false, bk0, (short)0, s[ni], false, false);
      s[ni] = __builtin_amdgcn_wmma_f32_16x16x32_bf16(false, aq[1].v, false, bk1, (short)0, s[ni], false, false);
    }

    // ---- scale + causal mask ----
#pragma unroll
    for (int ni = 0; ni < 2; ++ni) {
      const int key = kbase + ni * 16 + lr;
#pragma unroll
      for (int r = 0; r < 8; ++r) {
        const int row = q0 + lh * 8 + r;
        const float v = s[ni][r] * 0.125f;  // 1/sqrt(64)
        s[ni][r] = (key > row) ? -1e30f : v;
      }
    }

    // ---- online softmax (rows live in 16-lane halves of wave32) ----
#pragma unroll
    for (int r = 0; r < 8; ++r) {
      float t = fmaxf(s[0][r], s[1][r]);
#pragma unroll
      for (int msk = 8; msk >= 1; msk >>= 1) t = fmaxf(t, __shfl_xor(t, msk, 32));
      const float mnew = fmaxf(mrow[r], t);
      const float alpha = __expf(mrow[r] - mnew);
      const float p0 = __expf(s[0][r] - mnew);
      const float p1 = __expf(s[1][r] - mnew);
      float rs = p0 + p1;
#pragma unroll
      for (int msk = 8; msk >= 1; msk >>= 1) rs += __shfl_xor(rs, msk, 32);
      lrow[r] = lrow[r] * alpha + rs;
      mrow[r] = mnew;
#pragma unroll
      for (int j = 0; j < 4; ++j) accO[j][r] *= alpha;
      pstage[wave][lh * 8 + r][lr]      = (__bf16)p0;
      pstage[wave][lh * 8 + r][16 + lr] = (__bf16)p1;
    }

    // ---- reload P in A-fragment layout (same-wave LDS, in-order) ----
    Frag16 ap;
    ap.h[0] = *(const v8bf*)&pstage[wave][lr][lh * 16];
    ap.h[1] = *(const v8bf*)&pstage[wave][lr][lh * 16 + 8];

    // ---- O += P * V ----
#pragma unroll
    for (int j = 0; j < 4; ++j) {
      const __bf16* pv = Vb + (size_t)(j * 16 + lr) * T + kbase + lh * 16;
      v16bf bv = *(const v16bf*)pv;
      accO[j] = __builtin_amdgcn_wmma_f32_16x16x32_bf16(false, ap.v, false, bv, (short)0, accO[j], false, false);
    }
  }

  // ---- normalize + write [B*T, E] bf16 ----
#pragma unroll
  for (int j = 0; j < 4; ++j) {
#pragma unroll
    for (int r = 0; r < 8; ++r) {
      const int row = q0 + lh * 8 + r;
      const int col = h * 64 + j * 16 + lr;
      O[(size_t)((size_t)b * T + row) * E + col] = (__bf16)(accO[j][r] / lrow[r]);
    }
  }
}

// ---------------------------------------------------------------------------
// Host-side orchestration.
// ---------------------------------------------------------------------------
extern "C" void kernel_launch(void* const* d_in, const int* in_sizes, int n_in,
                              void* d_out, int out_size, void* d_ws, size_t ws_size,
                              hipStream_t stream) {
  (void)in_sizes; (void)n_in; (void)out_size; (void)ws_size;
  const int B = 2, T = 2048, E = 1024, H = 16, F = 4096;
  const int M = B * T;

  const float* x        = (const float*)d_in[0];
  const float* c_attn_w = (const float*)d_in[1];
  const float* c_attn_b = (const float*)d_in[2];
  const float* c_proj_w = (const float*)d_in[3];
  const float* c_proj_b = (const float*)d_in[4];
  const float* mlp_w1   = (const float*)d_in[5];
  const float* mlp_b1   = (const float*)d_in[6];
  const float* mlp_w2   = (const float*)d_in[7];
  const float* mlp_b2   = (const float*)d_in[8];
  const float* ln1_g    = (const float*)d_in[9];
  const float* ln1_bi   = (const float*)d_in[10];
  const float* ln2_g    = (const float*)d_in[11];
  const float* ln2_bi   = (const float*)d_in[12];
  float* out = (float*)d_out;

  char* ws = (char*)d_ws;
  size_t off = 0;
  auto bump = [&](size_t bytes) -> char* {
    char* p = ws + off;
    off += (bytes + 255) & ~(size_t)255;
    return p;
  };

  __bf16* wqkvT  = (__bf16*)bump((size_t)3 * E * E * 2);  // [3E][E]
  __bf16* wprojT = (__bf16*)bump((size_t)E * E * 2);      // [E][E]
  __bf16* w1T    = (__bf16*)bump((size_t)F * E * 2);      // [F][E]
  __bf16* w2T    = (__bf16*)bump((size_t)E * F * 2);      // [E][F]
  __bf16* ln1o   = (__bf16*)bump((size_t)M * E * 2);
  __bf16* qkvb   = (__bf16*)bump((size_t)M * 3 * E * 2);
  __bf16* Qc     = (__bf16*)bump((size_t)M * E * 2);
  __bf16* Kc     = (__bf16*)bump((size_t)M * E * 2);
  __bf16* Vt     = (__bf16*)bump((size_t)M * E * 2);
  __bf16* attnb  = (__bf16*)bump((size_t)M * E * 2);
  float*  xres   = (float* )bump((size_t)M * E * 4);
  __bf16* ln2o   = (__bf16*)bump((size_t)M * E * 2);
  __bf16* hb     = (__bf16*)bump((size_t)M * F * 2);

  const dim3 tb(32, 8);
  // Weights -> bf16, transposed to [N][K]
  transpose_to_bf16<<<dim3(3 * E / 32, E / 32), tb, 0, stream>>>(c_attn_w, wqkvT, E, 3 * E);
  transpose_to_bf16<<<dim3(E / 32,     E / 32), tb, 0, stream>>>(c_proj_w, wprojT, E, E);
  transpose_to_bf16<<<dim3(F / 32,     E / 32), tb, 0, stream>>>(mlp_w1,   w1T,    E, F);
  transpose_to_bf16<<<dim3(E / 32,     F / 32), tb, 0, stream>>>(mlp_w2,   w2T,    F, E);

  // LN1 -> QKV GEMM (bias fused) -> head split -> flash attention
  layernorm_to_bf16<<<M, 256, 0, stream>>>(x, ln1_g, ln1_bi, ln1o, E);
  gemm_bf16_wmma<EPI_BF16><<<dim3(M / 128, 3 * E / 64), 256, 0, stream>>>(
      ln1o, wqkvT, c_attn_b, nullptr, qkvb, nullptr, M, 3 * E, E);
  split_qkv<<<(M * E) / 256, 256, 0, stream>>>(qkvb, Qc, Kc, Vt, B, T, E, H);
  flash_attn<<<dim3(T / 128, H, B), 256, 0, stream>>>(Qc, Kc, Vt, attnb, T, H, E);

  // Out-proj GEMM with fused residual add (x + attn)
  gemm_bf16_wmma<EPI_RESF32><<<dim3(M / 128, E / 64), 256, 0, stream>>>(
      attnb, wprojT, c_proj_b, x, nullptr, xres, M, E, E);

  // LN2 -> MLP1 (fused GELU) -> MLP2 with fused residual -> final output
  layernorm_to_bf16<<<M, 256, 0, stream>>>(xres, ln2_g, ln2_bi, ln2o, E);
  gemm_bf16_wmma<EPI_GELU><<<dim3(M / 128, F / 64), 256, 0, stream>>>(
      ln2o, w1T, mlp_b1, nullptr, hb, nullptr, M, F, E);
  gemm_bf16_wmma<EPI_RESF32><<<dim3(M / 128, E / 64), 256, 0, stream>>>(
      hb, w2T, mlp_b2, xres, nullptr, out, M, E, F);
}
